// NLinear_57449482551297
// MI455X (gfx1250) — matrix operand order repaired and checked by reading, not compiled, not run on hardware
//
#include <hip/hip_runtime.h>
#include <stdint.h>

// out[b,f,o] = sum_i x[b,f,i] * w[f,i,o] + bias[f,o]
// B=4096, F=64, DIN=256, DOUT=256, fp32.
// Split-bf16 (hi+lo) GEMM: 3x v_wmma_f32_16x16x32_bf16 per tile => ~2^-16 rel accuracy,
// compute stays under the 23.3 TB/s HBM roofline floor (~24us for 554 MB of traffic).

typedef __attribute__((ext_vector_type(16))) __bf16 v16bf;
typedef __attribute__((ext_vector_type(8)))  float  v8f;

#define B_TOT 4096
#define F_TOT 64
#define DIN   256
#define DOUT  256

#define BM 128
#define BN 128
#define BK 32
#define KCHUNKS (DIN / BK)   // 8

// hi = bf16 truncation (returned packed, low half = first elem); residuals out-params.
__device__ __forceinline__ uint32_t pack_hi(float a, float b, float& ra, float& rb) {
    uint32_t ua = __float_as_uint(a) & 0xFFFF0000u;
    uint32_t ub = __float_as_uint(b) & 0xFFFF0000u;
    ra = a - __uint_as_float(ua);
    rb = b - __uint_as_float(ub);
    return (ua >> 16) | ub;
}
__device__ __forceinline__ uint32_t pack_lo(float ra, float rb) {
    return (__float_as_uint(ra) >> 16) | (__float_as_uint(rb) & 0xFFFF0000u);
}

__global__ __launch_bounds__(256)
void nlinear_wmma_bf16x3(const float* __restrict__ x,
                         const float* __restrict__ w,
                         const float* __restrict__ bias,
                         float* __restrict__ out) {
    // Double-buffered fragment-ready LDS: [buf][subtile(8)][lane(32)][8 dwords].
    __shared__ __align__(32) uint32_t AhL[2][2048];
    __shared__ __align__(32) uint32_t AlL[2][2048];
    __shared__ __align__(32) uint32_t BhL[2][2048];
    __shared__ __align__(32) uint32_t BlL[2][2048];

    const int tid  = threadIdx.x;
    const int lane = tid & 31;
    const int wave = tid >> 5;   // 0..7
    const int wm   = wave & 3;   // 4 waves along M (32 rows each)
    const int wn   = wave >> 2;  // 2 waves along N (64 cols each)

    const int f  = blockIdx.z;
    const int m0 = blockIdx.y * BM;
    const int n0 = blockIdx.x * BN;

    const float* xf = x + (size_t)f * DIN;                 // + row*F*DIN + k
    const float* wf = w + (size_t)f * DIN * DOUT;          // + k*DOUT + n

    // ---- staging geometry (constant per thread) ----
    // A: thread covers row (arow + 32*i), k-quad akq. 8 threads span one row's 32 floats.
    const int arow  = tid >> 3;            // 0..31
    const int akq   = (tid & 7) << 2;      // 0,4,...,28
    const int a_ms0 = arow >> 4;           // subtile grows by 2 per i
    const int a_q   = arow & 15;
    const int a_p   = (akq & 15) >> 3;
    const int a_vb  = ((akq & 7) >> 1) + ((akq >> 4) << 2);   // even => b64-aligned
    const int a_off0 = ((a_ms0 * 32) + (a_p * 16 + a_q)) * 8 + a_vb;

    // B: thread covers k-pair (bkp0 + 8*i), n-quad bn4. 32 threads span one row's 128 floats.
    const int bn4  = (tid & 31) << 2;      // 0..124
    const int bkp0 = tid >> 5;             // 0..7
    const int b_ns = bn4 >> 4;
    const int b_qb = bn4 & 15;

    v8f acc[2][4];
#pragma unroll
    for (int s = 0; s < 2; ++s)
#pragma unroll
        for (int t = 0; t < 4; ++t)
#pragma unroll
            for (int r = 0; r < 8; ++r) acc[s][t][r] = 0.0f;

    float4 pa[4];
    float4 pbe[2], pbo[2];

    auto issue_loads = [&](int k0) {
#pragma unroll
        for (int i = 0; i < 4; ++i)
            pa[i] = *(const float4*)(xf + (size_t)(m0 + arow + 32 * i) * (F_TOT * DIN) + (k0 + akq));
#pragma unroll
        for (int i = 0; i < 2; ++i) {
            const float* s = wf + (size_t)(k0 + 2 * bkp0 + 16 * i) * DOUT + (n0 + bn4);
            pbe[i] = *(const float4*)s;          // even-k row
            pbo[i] = *(const float4*)(s + DOUT); // odd-k row
        }
    };

    auto convert_store = [&](int buf) {
#pragma unroll
        for (int i = 0; i < 4; ++i) {
            float r0, r1, r2, r3;
            uint32_t h0 = pack_hi(pa[i].x, pa[i].y, r0, r1);
            uint32_t h1 = pack_hi(pa[i].z, pa[i].w, r2, r3);
            uint32_t l0 = pack_lo(r0, r1);
            uint32_t l1 = pack_lo(r2, r3);
            int off = a_off0 + i * 512;          // (+2 subtiles) * 32 lanes * 8 dwords
            *(uint2*)&AhL[buf][off] = make_uint2(h0, h1);
            *(uint2*)&AlL[buf][off] = make_uint2(l0, l1);
        }
#pragma unroll
        for (int i = 0; i < 2; ++i) {
            // k2 = 2*bkp0 + 16*i  =>  p = bkp0>>2, dword v = (bkp0&3) + 4*i
            float ev[4] = {pbe[i].x, pbe[i].y, pbe[i].z, pbe[i].w};
            float ov[4] = {pbo[i].x, pbo[i].y, pbo[i].z, pbo[i].w};
#pragma unroll
            for (int j = 0; j < 4; ++j) {
                float re, ro;
                uint32_t h = pack_hi(ev[j], ov[j], re, ro);
                uint32_t l = pack_lo(re, ro);
                int off = ((b_ns * 32) + ((bkp0 >> 2) * 16 + b_qb + j)) * 8 + (bkp0 & 3) + 4 * i;
                BhL[buf][off] = h;
                BlL[buf][off] = l;
            }
        }
    };

    // ---- prologue: chunk 0 ----
    issue_loads(0);
    convert_store(0);
    __syncthreads();

    for (int kc = 0; kc < KCHUNKS; ++kc) {
        const int buf = kc & 1;

        // Software pipeline: launch next chunk's global loads before the WMMA block.
        if (kc + 1 < KCHUNKS) issue_loads((kc + 1) * BK);

        const v16bf* Ahv = (const v16bf*)AhL[buf];
        const v16bf* Alv = (const v16bf*)AlL[buf];
        const v16bf* Bhv = (const v16bf*)BhL[buf];
        const v16bf* Blv = (const v16bf*)BlL[buf];

        v16bf ah[2], al[2], bh[4], bl[4];
#pragma unroll
        for (int s = 0; s < 2; ++s) {
            int ms = wm * 2 + s;
            ah[s] = Ahv[ms * 32 + lane];
            al[s] = Alv[ms * 32 + lane];
        }
#pragma unroll
        for (int t = 0; t < 4; ++t) {
            int ns = wn * 4 + t;
            bh[t] = Bhv[ns * 32 + lane];
            bl[t] = Blv[ns * 32 + lane];
        }

        // Term-major order: consecutive WMMAs never share an accumulator (no RAW chain).
#pragma unroll
        for (int s = 0; s < 2; ++s)
#pragma unroll
            for (int t = 0; t < 4; ++t)
                acc[s][t] = __builtin_amdgcn_wmma_f32_16x16x32_bf16(
                    false, ah[s], false, bh[t], (short)0, acc[s][t], false, false);
#pragma unroll
        for (int s = 0; s < 2; ++s)
#pragma unroll
            for (int t = 0; t < 4; ++t)
                acc[s][t] = __builtin_amdgcn_wmma_f32_16x16x32_bf16(
                    false, ah[s], false, bl[t], (short)0, acc[s][t], false, false);
#pragma unroll
        for (int s = 0; s < 2; ++s)
#pragma unroll
            for (int t = 0; t < 4; ++t)
                acc[s][t] = __builtin_amdgcn_wmma_f32_16x16x32_bf16(
                    false, al[s], false, bh[t], (short)0, acc[s][t], false, false);

        // Convert + store next chunk into the other buffer while this chunk's WMMAs drain.
        if (kc + 1 < KCHUNKS) convert_store(buf ^ 1);

        __syncthreads();
    }

    // ---- epilogue: bias + non-temporal streaming stores ----
    // C/D layout: element (r, lane): M = (lane/16)*8 + r, N = lane%16.
    const int qn = lane & 15;
    const int ph = lane >> 4;
#pragma unroll
    for (int t = 0; t < 4; ++t) {
        int n = n0 + wn * 64 + t * 16 + qn;
        float bv = bias[f * DOUT + n];
#pragma unroll
        for (int s = 0; s < 2; ++s) {
            int mbase = m0 + (wm * 2 + s) * 16 + ph * 8;
#pragma unroll
            for (int r = 0; r < 8; ++r) {
                size_t o = (size_t)(mbase + r) * (F_TOT * DOUT) + (size_t)f * DOUT + n;
                __builtin_nontemporal_store(acc[s][t][r] + bv, &out[o]);
            }
        }
    }
}

extern "C" void kernel_launch(void* const* d_in, const int* in_sizes, int n_in,
                              void* d_out, int out_size, void* d_ws, size_t ws_size,
                              hipStream_t stream) {
    const float* x    = (const float*)d_in[0];   // [4096, 64, 256]
    const float* wgt  = (const float*)d_in[1];   // [64, 256, 256]
    const float* bias = (const float*)d_in[2];   // [64, 256]
    float* out        = (float*)d_out;           // [4096, 64, 256]

    dim3 grid(DOUT / BN, B_TOT / BM, F_TOT);     // (2, 32, 64); n-tiles adjacent for L2 x-reuse
    dim3 block(256, 1, 1);
    nlinear_wmma_bf16x3<<<grid, block, 0, stream>>>(x, wgt, bias, out);
}
